// DecoderForLarge_40321152975399
// MI455X (gfx1250) — matrix operand order repaired and checked by reading, not compiled, not run on hardware
//
#include <hip/hip_runtime.h>
#include <hip/hip_bf16.h>

// ---------------------------------------------------------------------------
// Problem constants
// ---------------------------------------------------------------------------
#define B_  64
#define N_  1000
#define NP_ 1024     // N padded to multiple of 32 (bf16 WMMA K)
#define G_  500
#define GP_ 512      // G padded to multiple of 64 (g-tile)
#define H_  256

typedef __bf16 bf16_t;
typedef __attribute__((ext_vector_type(16))) __bf16 v16bf;
typedef __attribute__((ext_vector_type(8)))  __bf16 v8bf;
typedef __attribute__((ext_vector_type(8)))  float  v8f;

static __device__ __forceinline__ v16bf cat16(v8bf lo, v8bf hi) {
    return __builtin_shufflevector(lo, hi, 0,1,2,3,4,5,6,7,8,9,10,11,12,13,14,15);
}

// A-matrix (16x32 bf16) fragment: lane l<16 -> row M=l, K base 0 ; lanes 16-31 -> K base 8.
// elements 0..7 = K kb..kb+7, elements 8..15 = K kb+16..kb+23 (ISA 7.12.2).
static __device__ __forceinline__ v16bf loadA(const bf16_t* base, int rowStride, int lane, int k0) {
    const int r  = lane & 15;
    const int kb = k0 + ((lane < 16) ? 0 : 8);
    const bf16_t* p = base + (size_t)r * rowStride + kb;
    v8bf c0 = *(const v8bf*)(p);
    v8bf c1 = *(const v8bf*)(p + 16);
    return cat16(c0, c1);
}

// B-matrix (32x16 bf16) fragment: lane l holds column c=l&15, K kb..kb+7 and kb+16..kb+23 with
// kb = k0 + (l<16 ? 0 : 8). base points at element (K=0, col row-major with stride rowStride).
static __device__ __forceinline__ v16bf loadB(const bf16_t* base, int rowStride, int lane, int k0) {
    const int c  = lane & 15;
    const int kb = k0 + ((lane < 16) ? 0 : 8);
    const bf16_t* p = base + (size_t)c * rowStride + kb;
    v8bf c0 = *(const v8bf*)(p);
    v8bf c1 = *(const v8bf*)(p + 16);
    return cat16(c0, c1);
}

#define WMMA_BF16(a, b, c) \
    __builtin_amdgcn_wmma_f32_16x16x32_bf16(false, (a), false, (b), (short)0, (c), false, false)

// ---------------------------------------------------------------------------
// K1: embeddings f32 [B,N,H] -> embRM bf16 [B,NP,H] (row-major, zero-padded rows)
//                            -> embT  bf16 [B,H,NP] (transposed, zero-padded cols)
// ---------------------------------------------------------------------------
__global__ __launch_bounds__(256) void k_transpose(const float* __restrict__ emb,
                                                   bf16_t* __restrict__ embRM,
                                                   bf16_t* __restrict__ embT) {
    __shared__ float tile[64][65];
    const int hb = blockIdx.x * 64;
    const int nb = blockIdx.y * 64;
    const int b  = blockIdx.z;
    const int t  = threadIdx.x;
    const int c  = t & 63;
    const int r0 = t >> 6;            // 0..3
    #pragma unroll
    for (int i = 0; i < 16; ++i) {
        const int r = i * 4 + r0;     // 0..63
        const int n = nb + r;
        float v = 0.f;
        if (n < N_) v = emb[((size_t)b * N_ + n) * H_ + hb + c];
        tile[r][c] = v;
        embRM[((size_t)b * NP_ + n) * H_ + hb + c] = (bf16_t)v;
    }
    __syncthreads();
    #pragma unroll
    for (int i = 0; i < 16; ++i) {
        const int hh = i * 4 + r0;    // column within tile -> h
        embT[((size_t)b * H_ + hb + hh) * NP_ + nb + c] = (bf16_t)tile[c][hh];
    }
}

// ---------------------------------------------------------------------------
// K2: mean over N   -> mean f32 [B,H]
// ---------------------------------------------------------------------------
__global__ __launch_bounds__(256) void k_mean(const float* __restrict__ emb,
                                              float* __restrict__ mean) {
    const int b = blockIdx.x, h = threadIdx.x;
    const float* p = emb + (size_t)b * N_ * H_ + h;
    float s = 0.f;
    #pragma unroll 4
    for (int n = 0; n < N_; ++n) s += p[(size_t)n * H_];
    mean[b * H_ + h] = s * (1.f / (float)N_);
}

// K2b: q_graph[b,o] = sum_h mean[b,h] * Wq_graph[o,h]
__global__ __launch_bounds__(256) void k_qgraph(const float* __restrict__ mean,
                                                const float* __restrict__ wg,
                                                float* __restrict__ qgraph) {
    __shared__ float ms[H_];
    const int b = blockIdx.x, o = threadIdx.x;
    ms[o] = mean[b * H_ + o];
    __syncthreads();
    const float* wr = wg + (size_t)o * H_;
    float s = 0.f;
    #pragma unroll 8
    for (int h = 0; h < H_; ++h) s += ms[h] * wr[h];
    qgraph[b * H_ + o] = s;
}

// K2c: last-node embedding gather -> bf16 [B,GP,H] (pad rows zero)
__global__ __launch_bounds__(256) void k_lastemb(const float* __restrict__ emb,
                                                 const int* __restrict__ lastnode,
                                                 bf16_t* __restrict__ lastemb) {
    const int g = blockIdx.x, b = blockIdx.y, h = threadIdx.x;
    float v = 0.f;
    if (g < G_) {
        const int ln = lastnode[b * G_ + g];
        v = emb[((size_t)b * N_ + ln) * H_ + h];
    }
    lastemb[((size_t)b * GP_ + g) * H_ + h] = (bf16_t)v;
}

// Kw: fused/converted weights: Wlf = bf16(Wq_first + Wq_last), Wv = bf16(W_visited)
__global__ __launch_bounds__(256) void k_weights(const float* __restrict__ wq_first,
                                                 const float* __restrict__ wq_last,
                                                 const float* __restrict__ w_visited,
                                                 bf16_t* __restrict__ wlf,
                                                 bf16_t* __restrict__ wv) {
    const int i = blockIdx.x * 256 + threadIdx.x;
    wlf[i] = (bf16_t)(wq_first[i] + wq_last[i]);
    wv[i]  = (bf16_t)(w_visited[i]);
}

// ---------------------------------------------------------------------------
// K3: pooled[b,g,h] = (1/N) * sum_n binmask[b,g,n] * emb[b,n,h]   (bf16 WMMA)
// block: 64 g-rows x 256 h-cols, 8 waves (4 g-subtiles x 2 h-halves). K loop over n.
// Mask staging is branchless at chunk granularity: N_==1000 is a multiple of 8,
// and each staging thread owns an 8-aligned chunk, so chunks are fully in- or
// fully out-of-range -> two aligned float4 loads + cndmask converts + one
// ds_store_b128, no per-element EXEC juggling.
// ---------------------------------------------------------------------------
__global__ __launch_bounds__(256) void k_pooled(const float* __restrict__ mask,
                                                const bf16_t* __restrict__ embT,
                                                bf16_t* __restrict__ pooled) {
    __shared__ bf16_t At[64 * 40];    // 64 g-rows x 32 K (pad rows to 40 halves = 80B)
    __shared__ bf16_t Bt[256 * 40];   // 256 h-cols x 32 K
    const int b  = blockIdx.y;
    const int g0 = blockIdx.x * 64;
    const int tid = threadIdx.x, lane = tid & 31, w = tid >> 5;
    const int gsub = w & 3, hhalf = w >> 2;

    v8f acc[8];
    #pragma unroll
    for (int i = 0; i < 8; ++i) acc[i] = (v8f){0,0,0,0,0,0,0,0};

    const int gs = tid >> 2;          // staging: g row 0..63
    const int nq = (tid & 3) * 8;     // staging: n chunk (8-aligned)
    const int gg = g0 + gs;
    const float* mrow_p = mask + ((size_t)b * G_ + (gg < G_ ? gg : 0)) * N_;
    const bool rowok = (gg < G_);
    const bf16_t* bsrc_row = &embT[((size_t)b * H_ + tid) * NP_];

    for (int kt = 0; kt < NP_ / 32; ++kt) {
        const int n0 = kt * 32;
        __syncthreads();
        {   // stage A: binary mask -> bf16 (vectorized, branch only per 8-chunk)
            v8bf vv;
            #pragma unroll
            for (int j = 0; j < 8; ++j) vv[j] = (bf16_t)0.f;
            if (rowok && (n0 + nq) < N_) {
                const float4 f0 = *(const float4*)(mrow_p + n0 + nq);
                const float4 f1 = *(const float4*)(mrow_p + n0 + nq + 4);
                const float fa[8] = {f0.x, f0.y, f0.z, f0.w, f1.x, f1.y, f1.z, f1.w};
                #pragma unroll
                for (int j = 0; j < 8; ++j)
                    vv[j] = (__float_as_uint(fa[j]) == 0xFF800000u) ? (bf16_t)1.f
                                                                    : (bf16_t)0.f;
            }
            *(v8bf*)(&At[gs * 40 + nq]) = vv;
        }
        {   // stage B: embT row h=tid, 32 halves = 64B
            const bf16_t* src = bsrc_row + n0;
            bf16_t* dst = &Bt[tid * 40];
            *(v8bf*)(dst + 0)  = *(const v8bf*)(src + 0);
            *(v8bf*)(dst + 8)  = *(const v8bf*)(src + 8);
            *(v8bf*)(dst + 16) = *(const v8bf*)(src + 16);
            *(v8bf*)(dst + 24) = *(const v8bf*)(src + 24);
        }
        // prefetch next K-tile while this one is consumed
        if (kt + 1 < NP_ / 32) {
            if (rowok && (n0 + 32 + nq) < N_)
                __builtin_prefetch(mrow_p + n0 + 32 + nq, 0, 3);
            __builtin_prefetch(bsrc_row + n0 + 32, 0, 3);
        }
        __syncthreads();
        const v16bf a = loadA(&At[(gsub * 16) * 40], 40, lane, 0);
        #pragma unroll
        for (int t = 0; t < 8; ++t) {
            const v16bf bm = loadB(&Bt[(hhalf * 128 + t * 16) * 40], 40, lane, 0);
            acc[t] = WMMA_BF16(a, bm, acc[t]);
        }
    }
    const int mrow = (lane >> 4) * 8, col = lane & 15;
    #pragma unroll
    for (int t = 0; t < 8; ++t)
        #pragma unroll
        for (int r = 0; r < 8; ++r) {
            const int g = g0 + gsub * 16 + mrow + r;
            pooled[((size_t)b * GP_ + g) * H_ + hhalf * 128 + t * 16 + col] =
                (bf16_t)(acc[t][r] * (1.f / (float)N_));
        }
}

// ---------------------------------------------------------------------------
// K4: qfinal[b,g,o] = pooled@W_v^T + last_emb@W_lf^T + q_graph[b,o]   (-> bf16)
// Two WMMA chains share one accumulator.  B frags straight from global (L2 resident).
// ---------------------------------------------------------------------------
__global__ __launch_bounds__(256) void k_qfinal(const bf16_t* __restrict__ pooled,
                                                const bf16_t* __restrict__ lastemb,
                                                const bf16_t* __restrict__ wv,
                                                const bf16_t* __restrict__ wlf,
                                                const float* __restrict__ qgraph,
                                                bf16_t* __restrict__ qf) {
    const int b  = blockIdx.y;
    const int g0 = blockIdx.x * 64;
    const int tid = threadIdx.x, lane = tid & 31, w = tid >> 5;
    const int gsub = w & 3, ohalf = w >> 2;

    v8f acc[8];
    #pragma unroll
    for (int i = 0; i < 8; ++i) acc[i] = (v8f){0,0,0,0,0,0,0,0};

    const bf16_t* a1base = pooled  + ((size_t)b * GP_ + g0 + gsub * 16) * H_;
    const bf16_t* a2base = lastemb + ((size_t)b * GP_ + g0 + gsub * 16) * H_;
    for (int kt = 0; kt < H_ / 32; ++kt) {
        const int k0 = kt * 32;
        const v16bf a1 = loadA(a1base, H_, lane, k0);
        const v16bf a2 = loadA(a2base, H_, lane, k0);
        #pragma unroll
        for (int t = 0; t < 8; ++t) {
            const int o0 = ohalf * 128 + t * 16;
            const v16bf bv = loadB(wv  + (size_t)o0 * H_, H_, lane, k0);
            const v16bf bl = loadB(wlf + (size_t)o0 * H_, H_, lane, k0);
            acc[t] = WMMA_BF16(a1, bv, acc[t]);
            acc[t] = WMMA_BF16(a2, bl, acc[t]);
        }
    }
    const int mrow = (lane >> 4) * 8, col = lane & 15;
    #pragma unroll
    for (int t = 0; t < 8; ++t) {
        const int o = ohalf * 128 + t * 16 + col;
        const float qg = qgraph[b * H_ + o];
        #pragma unroll
        for (int r = 0; r < 8; ++r) {
            const int g = g0 + gsub * 16 + mrow + r;
            qf[((size_t)b * GP_ + g) * H_ + o] = (bf16_t)(acc[t][r] + qg);
        }
    }
}

// ---------------------------------------------------------------------------
// K5: score = qf@emb^T/sqrt(H) - dist/sqrt(2);  10*tanh; +mask_prob; softmax; store.
// block = 16 g-rows x full N (8 waves x 128 n each).  Softmax fused in-block.
// ---------------------------------------------------------------------------
__global__ __launch_bounds__(256) void k_score(const bf16_t* __restrict__ qf,
                                               const bf16_t* __restrict__ embRM,
                                               const float* __restrict__ dists,
                                               const float* __restrict__ mask,
                                               const int* __restrict__ lastnode,
                                               float* __restrict__ out) {
    __shared__ float redmax[16][8];
    __shared__ float redsum[16][8];
    __shared__ int   lnsh[16];
    const int b  = blockIdx.y;
    const int g0 = blockIdx.x * 16;
    const int tid = threadIdx.x, lane = tid & 31, w = tid >> 5;

    if (tid < 16) {
        const int g = g0 + tid;
        lnsh[tid] = (g < G_) ? lastnode[b * G_ + g] : 0;
    }

    v8f acc[8];
    #pragma unroll
    for (int i = 0; i < 8; ++i) acc[i] = (v8f){0,0,0,0,0,0,0,0};

    const bf16_t* abase = qf + ((size_t)b * GP_ + g0) * H_;
    for (int kt = 0; kt < H_ / 32; ++kt) {
        const int k0 = kt * 32;
        const v16bf a = loadA(abase, H_, lane, k0);
        #pragma unroll
        for (int t = 0; t < 8; ++t) {
            const bf16_t* bbase = embRM + ((size_t)b * NP_ + w * 128 + t * 16) * H_;
            const v16bf bm = loadB(bbase, H_, lane, k0);
            acc[t] = WMMA_BF16(a, bm, acc[t]);
        }
    }
    __syncthreads();   // lnsh visible

    const int mrow = (lane >> 4) * 8, col = lane & 15;
    float mx[8];
    #pragma unroll
    for (int r = 0; r < 8; ++r) mx[r] = -3.0e38f;

    #pragma unroll
    for (int t = 0; t < 8; ++t) {
        const int n = w * 128 + t * 16 + col;
        #pragma unroll
        for (int r = 0; r < 8; ++r) {
            const int g = g0 + mrow + r;
            float sc = -3.0e38f;
            if (n < N_ && g < G_) {
                const int ln = lnsh[mrow + r];
                const float d = dists[((size_t)b * N_ + ln) * N_ + n];
                const float s = acc[t][r] * 0.0625f - d * 0.70710678f;
                sc = 10.f * tanhf(s);
                const float m = mask[((size_t)b * G_ + g) * N_ + n];
                if (__float_as_uint(m) == 0xFF800000u) sc -= 1.0e8f;
            }
            acc[t][r] = sc;
            mx[r] = fmaxf(mx[r], sc);
        }
    }
    // reduce max across the 16 lanes holding the same rows
    #pragma unroll
    for (int off = 1; off < 16; off <<= 1)
        #pragma unroll
        for (int r = 0; r < 8; ++r) mx[r] = fmaxf(mx[r], __shfl_xor(mx[r], off, 32));
    if (col == 0) {
        #pragma unroll
        for (int r = 0; r < 8; ++r) redmax[mrow + r][w] = mx[r];
    }
    __syncthreads();
    float gmx[8];
    #pragma unroll
    for (int r = 0; r < 8; ++r) {
        float m = -3.0e38f;
        #pragma unroll
        for (int w2 = 0; w2 < 8; ++w2) m = fmaxf(m, redmax[mrow + r][w2]);
        gmx[r] = m;
    }
    // exponentiate + sum
    float sm[8];
    #pragma unroll
    for (int r = 0; r < 8; ++r) sm[r] = 0.f;
    #pragma unroll
    for (int t = 0; t < 8; ++t)
        #pragma unroll
        for (int r = 0; r < 8; ++r) {
            const float e = __expf(acc[t][r] - gmx[r]);
            acc[t][r] = e;
            sm[r] += e;
        }
    #pragma unroll
    for (int off = 1; off < 16; off <<= 1)
        #pragma unroll
        for (int r = 0; r < 8; ++r) sm[r] += __shfl_xor(sm[r], off, 32);
    if (col == 0) {
        #pragma unroll
        for (int r = 0; r < 8; ++r) redsum[mrow + r][w] = sm[r];
    }
    __syncthreads();
    float inv[8];
    #pragma unroll
    for (int r = 0; r < 8; ++r) {
        float s = 0.f;
        #pragma unroll
        for (int w2 = 0; w2 < 8; ++w2) s += redsum[mrow + r][w2];
        inv[r] = 1.f / s;
    }
    #pragma unroll
    for (int t = 0; t < 8; ++t) {
        const int n = w * 128 + t * 16 + col;
        if (n < N_) {
            #pragma unroll
            for (int r = 0; r < 8; ++r) {
                const int g = g0 + mrow + r;
                if (g < G_)
                    out[((size_t)b * G_ + g) * N_ + n] = acc[t][r] * inv[r];
            }
        }
    }
}

// ---------------------------------------------------------------------------
// Launch
// ---------------------------------------------------------------------------
extern "C" void kernel_launch(void* const* d_in, const int* in_sizes, int n_in,
                              void* d_out, int out_size, void* d_ws, size_t ws_size,
                              hipStream_t stream) {
    (void)in_sizes; (void)n_in; (void)out_size; (void)ws_size;
    const float* emb      = (const float*)d_in[0];   // [64,1000,256]
    const float* dists    = (const float*)d_in[1];   // [64,1000,1000]
    const int*   lastnode = (const int*)  d_in[2];   // [64,500]
    const float* mask     = (const float*)d_in[3];   // [64,500,1000] (0 or -inf)
    const float* wq_graph = (const float*)d_in[4];   // [256,256]
    const float* wq_first = (const float*)d_in[5];
    const float* wq_last  = (const float*)d_in[6];
    const float* w_visited= (const float*)d_in[7];
    float* out = (float*)d_out;                      // [64,500,1000]

    // workspace layout (all 256B-aligned)
    char* ws = (char*)d_ws;
    size_t off = 0;
    bf16_t* embT    = (bf16_t*)(ws + off); off += (size_t)B_ * H_  * NP_ * 2;  // 33.5 MB
    bf16_t* embRM   = (bf16_t*)(ws + off); off += (size_t)B_ * NP_ * H_  * 2;  // 33.5 MB
    bf16_t* pooled  = (bf16_t*)(ws + off); off += (size_t)B_ * GP_ * H_  * 2;  // 16.8 MB
    bf16_t* lastemb = (bf16_t*)(ws + off); off += (size_t)B_ * GP_ * H_  * 2;  // 16.8 MB
    bf16_t* qfin    = (bf16_t*)(ws + off); off += (size_t)B_ * GP_ * H_  * 2;  // 16.8 MB
    float*  qgraph  = (float*) (ws + off); off += (size_t)B_ * H_ * 4;
    float*  meanbuf = (float*) (ws + off); off += (size_t)B_ * H_ * 4;
    bf16_t* wlf     = (bf16_t*)(ws + off); off += (size_t)H_ * H_ * 2;
    bf16_t* wv      = (bf16_t*)(ws + off); off += (size_t)H_ * H_ * 2;

    k_weights  <<<dim3(H_ * H_ / 256), 256, 0, stream>>>(wq_first, wq_last, w_visited, wlf, wv);
    k_transpose<<<dim3(H_ / 64, NP_ / 64, B_), 256, 0, stream>>>(emb, embRM, embT);
    k_mean     <<<dim3(B_), 256, 0, stream>>>(emb, meanbuf);
    k_qgraph   <<<dim3(B_), 256, 0, stream>>>(meanbuf, wq_graph, qgraph);
    k_lastemb  <<<dim3(GP_, B_), 256, 0, stream>>>(emb, lastnode, lastemb);
    k_pooled   <<<dim3(GP_ / 64, B_), 256, 0, stream>>>(mask, embT, pooled);
    k_qfinal   <<<dim3(GP_ / 64, B_), 256, 0, stream>>>(pooled, lastemb, wv, wlf, qgraph, qfin);
    k_score    <<<dim3(GP_ / 16, B_), 256, 0, stream>>>(qfin, embRM, dists, mask, lastnode, out);
}
